// DiSA_14121852469285
// MI455X (gfx1250) — matrix operand order repaired and checked by reading, compiled, not run on hardware
//
#include <hip/hip_runtime.h>
#include <hip/hip_bf16.h>

typedef __bf16 bf16_t;
typedef __attribute__((ext_vector_type(16))) __bf16 v16bf;
typedef __attribute__((ext_vector_type(8)))  __bf16 v8bf;
typedef __attribute__((ext_vector_type(8)))  float  v8f;

#define LOG2E 1.4426950408889634f

// ---------------------------------------------------------------- converts --
__global__ __launch_bounds__(256) void cvt_f32_to_bf16(
    const float* __restrict__ in, bf16_t* __restrict__ out, int n) {
  int i = blockIdx.x * 256 + threadIdx.x;
  if (i < n) out[i] = (bf16_t)in[i];
}

// Transpose fp32 [K][N] weight -> bf16 [N][K] so B-fragments load contiguously.
__global__ __launch_bounds__(256) void transpose_to_bf16(
    const float* __restrict__ w, bf16_t* __restrict__ wT, int N) {
  __shared__ float tile[32][33];
  const int bx = blockIdx.x * 32, by = blockIdx.y * 32;
  const int tx = threadIdx.x & 31, ty = threadIdx.x >> 5;  // 8 rows / pass
  for (int r = ty; r < 32; r += 8)
    tile[r][tx] = w[(by + r) * N + bx + tx];
  __syncthreads();
  for (int r = ty; r < 32; r += 8)
    wT[(bx + r) * N + by + tx] = (bf16_t)tile[tx][r];
}

// ---------------------------------------------------------------- WMMA GEMM --
// C[m][n] = epilogue( sum_k A[m][k]*Bt[n][k] (+ sum_k A2[m][k]*Bt2[n][k]) + bias[n] )
// mode 0: ELU, write fp32 + bf16.  mode 1: linear, fp32.  mode 2: sigmoid gate blend.
// Per-wave 16x16 tile; K-step 32 via V_WMMA_F32_16X16X32_BF16.
// A frag (ISA 7.12.2): half=lane>>4, m=lane&15; 16 bf16 = K {half*8..+7, half*8+16..+23}
// B frag: n=lane&15; 16 bf16 = K {half*8..+7, half*8+16..+23} -> two b128 loads.
__global__ __launch_bounds__(128) void gemm_wmma_bf16(
    const bf16_t* __restrict__ A,  const bf16_t* __restrict__ Bt,
    const bf16_t* __restrict__ A2, const bf16_t* __restrict__ Bt2,
    const float* __restrict__ bias,
    float* __restrict__ outF, bf16_t* __restrict__ outB,
    const float* __restrict__ rep, const float* __restrict__ attn,
    int N, int K, int mode) {
  const int wave = threadIdx.x >> 5;
  const int lane = threadIdx.x & 31;
  const int half = lane >> 4;
  const int l15  = lane & 15;
  const int m0 = blockIdx.x * 16;
  const int n0 = (blockIdx.y * 4 + wave) * 16;

  v8f acc = {};
  const long arow = (long)(m0 + l15) * K + half * 8;
  const long brow = (long)(n0 + l15) * K + half * 8;
  for (int k0 = 0; k0 < K; k0 += 32) {
    union { v16bf v; v8bf h[2]; } a, b;
    a.h[0] = *(const v8bf*)(A + arow + k0);
    a.h[1] = *(const v8bf*)(A + arow + k0 + 16);
    b.h[0] = *(const v8bf*)(Bt + brow + k0);
    b.h[1] = *(const v8bf*)(Bt + brow + k0 + 16);
    acc = __builtin_amdgcn_wmma_f32_16x16x32_bf16(
        false, a.v, false, b.v, (short)0, acc, false, false);
  }
  if (A2 != nullptr) {  // second accumulation pair (gate: attn @ wf2)
    for (int k0 = 0; k0 < K; k0 += 32) {
      union { v16bf v; v8bf h[2]; } a, b;
      a.h[0] = *(const v8bf*)(A2 + arow + k0);
      a.h[1] = *(const v8bf*)(A2 + arow + k0 + 16);
      b.h[0] = *(const v8bf*)(Bt2 + brow + k0);
      b.h[1] = *(const v8bf*)(Bt2 + brow + k0 + 16);
      acc = __builtin_amdgcn_wmma_f32_16x16x32_bf16(
          false, a.v, false, b.v, (short)0, acc, false, false);
    }
  }

  const int n  = n0 + l15;
  const float bv = bias[n];
#pragma unroll
  for (int r = 0; r < 8; ++r) {
    const int m = m0 + half * 8 + r;  // C/D layout: VGPR r -> row r (lanes0-15) / r+8
    const long idx = (long)m * N + n;
    float v = acc[r] + bv;
    if (mode == 0) {            // ELU
      float e = v > 0.f ? v : __builtin_amdgcn_exp2f(v * LOG2E) - 1.f;
      outF[idx] = e;
      outB[idx] = (bf16_t)e;
    } else if (mode == 1) {     // linear
      outF[idx] = v;
    } else {                    // sigmoid gate + blend
      float g = __builtin_amdgcn_rcpf(1.f + __builtin_amdgcn_exp2f(-v * LOG2E));
      float rv = rep[idx], av = attn[idx];
      outF[idx] = g * rv + (1.f - g) * av;
    }
  }
}

// ------------------------------------------------------- fused DiSA attention --
// One block per (b, i); thread = 2 channels (float2). Softmax over dependent j.
// Identity used:  exp(5*tanh(u/5) - 5) = exp(-10 / (1 + e^{0.4u}))
// Logits bounded in [-5,5] -> fixed stabilizer 5, no online max needed.
// Masked entries (j <= i, i < L-1) underflow to exactly 0 vs max >= -5: skip them.
// Row i == L-1 is fully masked: the -1e9 shift cancels, loop all j unmasked.
__global__ __launch_bounds__(256) void disa_attn(
    const float* __restrict__ rep, const float* __restrict__ dep,
    const float* __restrict__ head, const float* __restrict__ blog,
    float* __restrict__ attnF, bf16_t* __restrict__ attnB, int L, int H) {
  const int b = blockIdx.y;
  const int i = blockIdx.x;
  const int h = threadIdx.x * 2;
  const long rowi = ((long)b * L + i) * H + h;
  const float2 hd = *(const float2*)(head + rowi);
  const float2 bl = *(const float2*)(blog + h);
  const float hv0 = hd.x + bl.x;
  const float hv1 = hd.y + bl.y;
  const float* depb = dep + (long)b * L * H + h;
  const float* repb = rep + (long)b * L * H + h;
  const int jstart = (i == L - 1) ? 0 : i + 1;

  const float K1 = 0.4f * LOG2E;     // exponent scale for e^{0.4u} via exp2
  const float K2 = -10.f * LOG2E;    // exp(-10*q) via exp2
  float s0 = 0.f, num0 = 0.f;
  float s1 = 0.f, num1 = 0.f;
  for (int j = jstart; j < L; ++j) {
    const float2 d = *(const float2*)(depb + (long)j * H);
    const float2 r = *(const float2*)(repb + (long)j * H);
    {
      float e = __builtin_amdgcn_exp2f((d.x + hv0) * K1);
      float q = __builtin_amdgcn_rcpf(1.f + e);
      float p = __builtin_amdgcn_exp2f(K2 * q);   // = exp(logit - 5) in (0, 1]
      s0 += p;  num0 += p * r.x;
    }
    {
      float e = __builtin_amdgcn_exp2f((d.y + hv1) * K1);
      float q = __builtin_amdgcn_rcpf(1.f + e);
      float p = __builtin_amdgcn_exp2f(K2 * q);
      s1 += p;  num1 += p * r.y;
    }
  }
  const float o0 = num0 * __builtin_amdgcn_rcpf(s0);
  const float o1 = num1 * __builtin_amdgcn_rcpf(s1);
  attnF[rowi]     = o0;
  attnF[rowi + 1] = o1;
  attnB[rowi]     = (bf16_t)o0;
  attnB[rowi + 1] = (bf16_t)o1;
}

// ------------------------------------------------------------------- launch --
extern "C" void kernel_launch(void* const* d_in, const int* in_sizes, int n_in,
                              void* d_out, int out_size, void* d_ws, size_t ws_size,
                              hipStream_t stream) {
  const float* x       = (const float*)d_in[0];
  const float* fc_w    = (const float*)d_in[1];
  const float* fc_b    = (const float*)d_in[2];
  const float* w1_w    = (const float*)d_in[3];
  const float* w1_b    = (const float*)d_in[4];
  const float* w2_w    = (const float*)d_in[5];
  const float* w2_b    = (const float*)d_in[6];
  const float* b_logit = (const float*)d_in[7];
  const float* wf1_w   = (const float*)d_in[8];
  const float* wf2_w   = (const float*)d_in[9];
  const float* bfv     = (const float*)d_in[10];
  float* out = (float*)d_out;

  const int B = 2, L = 256, H = 512, K = 512, N = 512;
  const int M = B * L;  // 512

  char* ws = (char*)d_ws;
  size_t off = 0;
  auto carve = [&](size_t bytes) {
    void* p = ws + off;
    off += (bytes + 255) & ~(size_t)255;
    return p;
  };
  bf16_t* xb   = (bf16_t*)carve((size_t)M * K * 2);
  bf16_t* tfc  = (bf16_t*)carve((size_t)K * N * 2);
  bf16_t* tw1  = (bf16_t*)carve((size_t)K * N * 2);
  bf16_t* tw2  = (bf16_t*)carve((size_t)K * N * 2);
  bf16_t* tf1  = (bf16_t*)carve((size_t)K * N * 2);
  bf16_t* tf2  = (bf16_t*)carve((size_t)K * N * 2);
  bf16_t* repB = (bf16_t*)carve((size_t)M * N * 2);
  bf16_t* attB = (bf16_t*)carve((size_t)M * N * 2);
  float*  repF = (float*)carve((size_t)M * N * 4);
  float*  depF = (float*)carve((size_t)M * N * 4);
  float*  hedF = (float*)carve((size_t)M * N * 4);
  float*  attF = (float*)carve((size_t)M * N * 4);

  // Stage 0: precision conversion + weight transposes (bf16 [N][K])
  cvt_f32_to_bf16<<<(M * K + 255) / 256, 256, 0, stream>>>(x, xb, M * K);
  dim3 tg(N / 32, K / 32);
  transpose_to_bf16<<<tg, 256, 0, stream>>>(fc_w,  tfc, N);
  transpose_to_bf16<<<tg, 256, 0, stream>>>(w1_w,  tw1, N);
  transpose_to_bf16<<<tg, 256, 0, stream>>>(w2_w,  tw2, N);
  transpose_to_bf16<<<tg, 256, 0, stream>>>(wf1_w, tf1, N);
  transpose_to_bf16<<<tg, 256, 0, stream>>>(wf2_w, tf2, N);

  dim3 gg(M / 16, N / 64);  // 4 waves/block, 16x16 tile per wave
  // Stage 1: rep = elu(x @ fc_w + fc_b)   (fp32 + bf16 copies)
  gemm_wmma_bf16<<<gg, 128, 0, stream>>>(xb, tfc, nullptr, nullptr, fc_b,
                                         repF, repB, nullptr, nullptr, N, K, 0);
  // Stage 2: dependent / head projections
  gemm_wmma_bf16<<<gg, 128, 0, stream>>>(repB, tw1, nullptr, nullptr, w1_b,
                                         depF, nullptr, nullptr, nullptr, N, K, 1);
  gemm_wmma_bf16<<<gg, 128, 0, stream>>>(repB, tw2, nullptr, nullptr, w2_b,
                                         hedF, nullptr, nullptr, nullptr, N, K, 1);
  // Stage 3: fused masked tanh-logit softmax attention (never materialize LxLxH)
  dim3 ag(L, B);
  disa_attn<<<ag, H / 2, 0, stream>>>(repF, depF, hedF, b_logit, attF, attB, L, H);
  // Stage 4: gate = sigmoid(rep@wf1 + attn@wf2 + bf); out = g*rep + (1-g)*attn
  gemm_wmma_bf16<<<gg, 128, 0, stream>>>(repB, tf1, attB, tf2, bfv,
                                         out, nullptr, repF, attF, N, K, 2);
}
